// GroupedQueryAttention_30631706755355
// MI455X (gfx1250) — matrix operand order repaired and checked by reading, not compiled, not run on hardware
//
#include <hip/hip_runtime.h>
#include <hip/hip_bf16.h>

typedef __attribute__((ext_vector_type(16))) __bf16 v16bf;
typedef __attribute__((ext_vector_type(8)))  float  v8f;
typedef __attribute__((ext_vector_type(4)))  unsigned int u32x4;
typedef __attribute__((ext_vector_type(8)))  unsigned int u32x8;

union FragB16 { u32x4 q[2]; v16bf v; };

// ---------------------------------------------------------------------------
// Tensor Data Mover: 2D tile load global -> LDS (D# groups per ISA 08 §8.3/8.4)
// dszCode: 0=1B,1=2B,2=4B. td0 = tile width (elems), td1 = tile rows,
// stride0 = row stride in elems. 2-SGPR-group form (2D tensor, VADDR2/3=NULL).
// ---------------------------------------------------------------------------
__device__ __forceinline__ void tdm_load_2d(unsigned ldsOff, unsigned long long ga,
                                            unsigned dszCode, unsigned td0, unsigned td1,
                                            unsigned long long stride0) {
  u32x4 g0;
  g0[0] = 1u;                                               // count=1 (valid descriptor)
  g0[1] = ldsOff;                                           // lds_addr
  g0[2] = (unsigned)(ga & 0xffffffffull);                   // global_addr[31:0]
  g0[3] = (unsigned)((ga >> 32) & 0x1ffffffull) | (2u << 30); // addr[56:32] | type=2
  unsigned long long q0 = ((unsigned long long)dszCode << 16) |
                          ((unsigned long long)(td0 & 0xffffu) << 48);   // tensor_dim0 lo
  unsigned long long q1 = ((unsigned long long)(td0 >> 16)) |           // tensor_dim0 hi
                          ((unsigned long long)td1 << 16) |             // tensor_dim1
                          ((unsigned long long)(td0 & 0xffffu) << 48);  // tile_dim0
  unsigned long long q2 = ((unsigned long long)(td1 & 0xffffu)) |       // tile_dim1
                          ((stride0 & 0xffffffffull) << 32);            // dim0_stride lo
  unsigned long long q3 = (stride0 >> 32) & 0xffffull;                  // dim0_stride hi
  u32x8 g1;
  g1[0] = (unsigned)q0; g1[1] = (unsigned)(q0 >> 32);
  g1[2] = (unsigned)q1; g1[3] = (unsigned)(q1 >> 32);
  g1[4] = (unsigned)q2; g1[5] = (unsigned)(q2 >> 32);
  g1[6] = (unsigned)q3; g1[7] = (unsigned)(q3 >> 32);
  asm volatile("tensor_load_to_lds %0, %1" :: "s"(g0), "s"(g1) : "memory");
}

// ---------------------------------------------------------------------------
// WMMA fragment builders from LDS tiles.
// A (16x32 bf16, M striped over lanes): lane<16 row=lane holds K{0..7,16..23},
// lane>=16 row=lane-16 holds K{8..15,24..31}.  LDS A tile is [rows][32] row-major.
// ---------------------------------------------------------------------------
template<bool A_F32>
__device__ __forceinline__ v16bf load_frag_a(const char* lds, int row, int lane) {
  const int r  = row + (lane & 15);
  const int hi = lane >> 4;
  FragB16 f;
  if constexpr (!A_F32) {
    const char* p = lds + r * 64 + hi * 16;          // 32 elems * 2B per row
    f.q[0] = *(const u32x4*)(p);                     // K0..7 | K8..15
    f.q[1] = *(const u32x4*)(p + 32);                // K16..23 | K24..31
  } else {
    const float* p = (const float*)(lds + r * 128) + hi * 8;  // 32 elems * 4B per row
    float t[16];
#pragma unroll
    for (int e = 0; e < 8; ++e) t[e] = p[e];
#pragma unroll
    for (int e = 0; e < 8; ++e) t[8 + e] = p[16 + e];
#pragma unroll
    for (int e = 0; e < 16; ++e) f.v[e] = (__bf16)t[e];
  }
  return f.v;
}

// B (32x16 bf16, N striped over lanes): lane<16 col=lane holds K{0..7,16..23},
// lane>=16 col=lane-16 holds K{8..15,24..31}.  LDS B tile is [cols][32] row-major (Bᵀ).
__device__ __forceinline__ v16bf load_frag_b(const char* lds, int col, int lane) {
  const int c  = col + (lane & 15);
  const int hi = lane >> 4;
  const char* p = lds + c * 64 + hi * 16;
  FragB16 f;
  f.q[0] = *(const u32x4*)(p);
  f.q[1] = *(const u32x4*)(p + 32);
  return f.v;
}

// ---------------------------------------------------------------------------
// Batched NT GEMM: C[z] = A[z] (MxK) * B[z>>bShift] (NxK)^T, bf16 WMMA, f32 acc.
// Waves tiled WRxWC over a (WR*64)x(WC*64) block tile; K stepped by 32 with
// TDM double-buffered LDS staging.
// ---------------------------------------------------------------------------
template<int WR, int WC, bool A_F32, bool OUT_BF16>
__global__ __launch_bounds__(WR * WC * 32) void
gemm_nt(const void* __restrict__ Abase, const __bf16* __restrict__ Bbase,
        void* __restrict__ Cbase,
        int M, int N, int K, int lda, int ldb, int ldc,
        long long batchA, long long batchB, long long batchC, int bShift,
        float scale) {
  (void)M; (void)N;
  constexpr int TM = WR * 64, TN = WC * 64, KSTEP = 32;
  constexpr int ABYTES = A_F32 ? 4 : 2;
  __shared__ __align__(16) char ldsA[2][TM * KSTEP * ABYTES];
  __shared__ __align__(16) char ldsB[2][TN * KSTEP * 2];

  const int z = blockIdx.z;
  const char*   Ab = (const char*)Abase + (long long)z * batchA * ABYTES;
  const __bf16* Bb = Bbase + (long long)(z >> bShift) * batchB;

  const int rowBase = blockIdx.y * TM;
  const int colBase = blockIdx.x * TN;
  const int lane = threadIdx.x & 31;
  const int wave = threadIdx.x >> 5;
  const int wRow = (wave / WC) * 64;
  const int wCol = (wave % WC) * 64;

  const v8f vzero = {0.f, 0.f, 0.f, 0.f, 0.f, 0.f, 0.f, 0.f};
  v8f acc[4][4];
#pragma unroll
  for (int i = 0; i < 4; ++i)
#pragma unroll
    for (int j = 0; j < 4; ++j) acc[i][j] = vzero;

  const int nk = K / KSTEP;
  auto issue = [&](int buf, int kt) {
    unsigned long long ga = (unsigned long long)Ab +
        ((long long)rowBase * lda + (long long)kt * KSTEP) * ABYTES;
    tdm_load_2d((unsigned)(size_t)&ldsA[buf][0], ga, A_F32 ? 2u : 1u,
                KSTEP, TM, (unsigned long long)lda);
    unsigned long long gb = (unsigned long long)(Bb + (long long)colBase * ldb +
                                                 (long long)kt * KSTEP);
    tdm_load_2d((unsigned)(size_t)&ldsB[buf][0], gb, 1u,
                KSTEP, TN, (unsigned long long)ldb);
  };

  if (wave == 0) issue(0, 0);
  for (int kt = 0; kt < nk; ++kt) {
    const int cur = kt & 1;
    if (wave == 0) {
      if (kt + 1 < nk) {
        issue(cur ^ 1, kt + 1);                      // prefetch next buffer
        __builtin_amdgcn_s_wait_tensorcnt(2);        // current pair complete
      } else {
        __builtin_amdgcn_s_wait_tensorcnt(0);
      }
    }
    __syncthreads();                                 // tile ready for all waves

    v16bf aF[4], bF[4];
#pragma unroll
    for (int i = 0; i < 4; ++i) aF[i] = load_frag_a<A_F32>(ldsA[cur], wRow + i * 16, lane);
#pragma unroll
    for (int j = 0; j < 4; ++j) bF[j] = load_frag_b(ldsB[cur], wCol + j * 16, lane);
#pragma unroll
    for (int i = 0; i < 4; ++i)
#pragma unroll
      for (int j = 0; j < 4; ++j)
        acc[i][j] = __builtin_amdgcn_wmma_f32_16x16x32_bf16(
            false, aF[i], false, bF[j], (short)0, acc[i][j], false, false);

    __syncthreads();                                 // done reading before rewrite
  }

  // C/D layout: lane<16 -> (M=v, N=lane); lane>=16 -> (M=v+8, N=lane-16)
  const int r0 = rowBase + wRow + ((lane >> 4) << 3);
  const int c0 = colBase + wCol + (lane & 15);
  if (OUT_BF16) {
    __bf16* C = (__bf16*)Cbase + (long long)z * batchC;
#pragma unroll
    for (int i = 0; i < 4; ++i)
#pragma unroll
      for (int j = 0; j < 4; ++j)
#pragma unroll
        for (int v = 0; v < 8; ++v)
          C[(long long)(r0 + i * 16 + v) * ldc + (c0 + j * 16)] =
              (__bf16)(acc[i][j][v] * scale);
  } else {
    float* C = (float*)Cbase + (long long)z * batchC;
#pragma unroll
    for (int i = 0; i < 4; ++i)
#pragma unroll
      for (int j = 0; j < 4; ++j)
#pragma unroll
        for (int v = 0; v < 8; ++v)
          C[(long long)(r0 + i * 16 + v) * ldc + (c0 + j * 16)] =
              acc[i][j][v] * scale;
  }
}

// ---------------------------------------------------------------------------
// Elementwise fp32 -> bf16
// ---------------------------------------------------------------------------
__global__ void cvt_bf16(const float* __restrict__ in, __bf16* __restrict__ out, int n) {
  int i = (blockIdx.x * blockDim.x + threadIdx.x) * 4;
  if (i + 3 < n) {
    float4 v = *(const float4*)(in + i);
    out[i + 0] = (__bf16)v.x; out[i + 1] = (__bf16)v.y;
    out[i + 2] = (__bf16)v.z; out[i + 3] = (__bf16)v.w;
  }
}

// fp32 [K,N] -> bf16 [N,K] (LDS-tiled transpose)
__global__ __launch_bounds__(256) void transpose_cvt(const float* __restrict__ in,
                                                     __bf16* __restrict__ outT,
                                                     int K, int N) {
  __shared__ float tile[32][33];
  const int n0 = blockIdx.x * 32, k0 = blockIdx.y * 32;
  const int tx = threadIdx.x, ty = threadIdx.y;
#pragma unroll
  for (int r = 0; r < 32; r += 8)
    tile[ty + r][tx] = in[(size_t)(k0 + ty + r) * N + (n0 + tx)];
  __syncthreads();
#pragma unroll
  for (int r = 0; r < 32; r += 8)
    outT[(size_t)(n0 + ty + r) * K + (k0 + tx)] = (__bf16)tile[tx][ty + r];
}

// In-place row softmax over S=2048 columns; one 256-thread block per row.
__global__ __launch_bounds__(256) void softmax_row(float* __restrict__ attn) {
  const int S = 2048;
  float* row = attn + (size_t)blockIdx.x * S;
  const int t = threadIdx.x;
  float vals[8];
  float m = -3.402823466e38f;
#pragma unroll
  for (int i = 0; i < 8; ++i) { vals[i] = row[t + i * 256]; m = fmaxf(m, vals[i]); }
  __shared__ float red[256];
  red[t] = m; __syncthreads();
  for (int s = 128; s > 0; s >>= 1) { if (t < s) red[t] = fmaxf(red[t], red[t + s]); __syncthreads(); }
  m = red[0]; __syncthreads();
  float sum = 0.f;
#pragma unroll
  for (int i = 0; i < 8; ++i) { vals[i] = __expf(vals[i] - m); sum += vals[i]; }
  red[t] = sum; __syncthreads();
  for (int s = 128; s > 0; s >>= 1) { if (t < s) red[t] += red[t + s]; __syncthreads(); }
  const float inv = 1.0f / red[0];
#pragma unroll
  for (int i = 0; i < 8; ++i) row[t + i * 256] = vals[i] * inv;
}

// ---------------------------------------------------------------------------
extern "C" void kernel_launch(void* const* d_in, const int* in_sizes, int n_in,
                              void* d_out, int out_size, void* d_ws, size_t ws_size,
                              hipStream_t stream) {
  (void)in_sizes; (void)n_in; (void)out_size; (void)ws_size;
  constexpr int S = 2048, D = 2048, H = 32, HD = 64, KVD = 512; // KVH*HD

  const float* x  = (const float*)d_in[0];
  const float* Wq = (const float*)d_in[1];
  const float* Wk = (const float*)d_in[2];
  const float* Wv = (const float*)d_in[3];
  const float* Wo = (const float*)d_in[4];
  float* out  = (float*)d_out;                 // [S, D]
  float* attn = out + (size_t)S * D;           // [H, S, S]

  __bf16* p   = (__bf16*)d_ws;
  __bf16* xb  = p; p += (size_t)S * D;         // x bf16 [S,D]
  __bf16* WqT = p; p += (size_t)D * D;         // Wqᵀ [D,D]
  __bf16* WkT = p; p += (size_t)KVD * D;       // Wkᵀ [KVD,D]
  __bf16* WvT = p; p += (size_t)KVD * D;       // Wvᵀ [KVD,D]
  __bf16* WoT = p; p += (size_t)D * D;         // Woᵀ [D,D]
  __bf16* Qb  = p; p += (size_t)S * D;         // Q [S,D]
  __bf16* Kb  = p; p += (size_t)S * KVD;       // K [S,KVD]
  __bf16* Vt  = p; p += (size_t)KVD * S;       // Vᵀ [KVD,S]  (= [KVH][HD][S])
  __bf16* Ob  = p;                             // attn·V concat [S,D]

  cvt_bf16<<<(S * D / 4 + 255) / 256, 256, 0, stream>>>(x, xb, S * D);
  dim3 tb(32, 8);
  transpose_cvt<<<dim3(D / 32,   D / 32), tb, 0, stream>>>(Wq, WqT, D, D);
  transpose_cvt<<<dim3(KVD / 32, D / 32), tb, 0, stream>>>(Wk, WkT, D, KVD);
  transpose_cvt<<<dim3(KVD / 32, D / 32), tb, 0, stream>>>(Wv, WvT, D, KVD);
  transpose_cvt<<<dim3(D / 32,   D / 32), tb, 0, stream>>>(Wo, WoT, D, D);

  // Q = x·Wq   (NT with pre-transposed weights)
  gemm_nt<2, 2, false, true><<<dim3(D / 128, S / 128, 1), 128, 0, stream>>>(
      xb, WqT, Qb, S, D, D, D, D, D, 0, 0, 0, 0, 1.0f);
  // K = x·Wk
  gemm_nt<2, 2, false, true><<<dim3(KVD / 128, S / 128, 1), 128, 0, stream>>>(
      xb, WkT, Kb, S, KVD, D, D, D, KVD, 0, 0, 0, 0, 1.0f);
  // Vᵀ = Wvᵀ·xᵀ  -> directly transposed V for the attn·V NT GEMM
  gemm_nt<2, 2, false, true><<<dim3(S / 128, KVD / 128, 1), 128, 0, stream>>>(
      WvT, xb, Vt, KVD, S, D, D, D, S, 0, 0, 0, 0, 1.0f);

  // scores[h] = Q_h · K_{h/4}ᵀ / 8   (fp32 into d_out attn region)
  gemm_nt<2, 2, false, false><<<dim3(S / 128, S / 128, H), 128, 0, stream>>>(
      Qb, Kb, attn, S, S, HD, D, KVD, S,
      /*batchA=*/HD, /*batchB=*/HD, /*batchC=*/(long long)S * S, /*bShift=*/2, 0.125f);

  softmax_row<<<H * S, 256, 0, stream>>>(attn);

  // out_h = attn_h · V_{h/4}   (A read as fp32 from d_out, converted in-register)
  gemm_nt<2, 1, true, true><<<dim3(HD / 64, S / 128, H), 64, 0, stream>>>(
      attn, Vt, Ob, S, HD, S, S, S, D,
      /*batchA=*/(long long)S * S, /*batchB=*/(long long)HD * S, /*batchC=*/HD,
      /*bShift=*/2, 1.0f);

  // output = concat(out_h) · Wo
  gemm_nt<2, 2, false, false><<<dim3(D / 128, S / 128, 1), 128, 0, stream>>>(
      Ob, WoT, out, S, D, D, D, D, D, 0, 0, 0, 0, 1.0f);
}
